// SemanticClipAttnProcessor_5617817223517
// MI455X (gfx1250) — compile-verified, matrix operand order checked by reading, not run on hardware
//
#include <hip/hip_runtime.h>

#define HEADS   20
#define DH      64
#define S_LEN   4096
#define C_DIM   1280
#define CD_DIM  2048
#define TXT     77
#define NTOK    4
#define BATCH   4

typedef __attribute__((ext_vector_type(16))) __bf16 v16bf;
typedef __attribute__((ext_vector_type(8)))  float  v8f;

union FragU {
    v16bf v;
    unsigned short u[16];
    uint4 q[2];
};

__device__ __forceinline__ unsigned short f2bf(float f) {
    union { float f; unsigned int u; } x; x.f = f;
    unsigned int r = x.u + 0x7fffu + ((x.u >> 16) & 1u);   // round-to-nearest-even
    return (unsigned short)(r >> 16);
}
__device__ __forceinline__ float bf2f(unsigned short h) {
    union { unsigned int u; float f; } x; x.u = ((unsigned int)h) << 16;
    return x.f;
}

// 16-bit A/B fragment K mapping (CDNA5 ISA 7.12.2):
// lanes 0-15 hold K = {0..7, 16..23}, lanes 16-31 hold K = {8..15, 24..31}
__device__ __forceinline__ int frag_k(int e, int g) {
    return (e < 8) ? (g * 8 + e) : (16 + g * 8 + (e - 8));
}

// Fragment whose 16 bf16 elements are contiguous in memory along K:
// two 16-byte loads (K = g*8 + [0..7] and K = 16 + g*8 + [0..7]).
__device__ __forceinline__ v16bf load_frag_contig(const unsigned short* p, int g) {
    FragU f;
    f.q[0] = *(const uint4*)(p + g * 8);
    f.q[1] = *(const uint4*)(p + 16 + g * 8);
    return f.v;
}

#define WMMA_BF16(a, b, c) \
    __builtin_amdgcn_wmma_f32_16x16x32_bf16(false, (a), false, (b), (short)0, (c), false, false)

// ---------------------------------------------------------------- conversions
__global__ void cvt_bf16_kernel(const float* __restrict__ x,
                                unsigned short* __restrict__ y, long long n) {
    long long i = (long long)blockIdx.x * blockDim.x + threadIdx.x;
    long long stride = (long long)gridDim.x * blockDim.x;
    for (; i < n; i += stride) y[i] = f2bf(x[i]);
}

// W[K][N] fp32 -> Wt[N][K] bf16 (col-major weights => contiguous-K B fragments)
__global__ void cvt_t_bf16_kernel(const float* __restrict__ w,
                                  unsigned short* __restrict__ wt, int K, int N) {
    long long total = (long long)K * N;
    long long i = (long long)blockIdx.x * blockDim.x + threadIdx.x;
    long long stride = (long long)gridDim.x * blockDim.x;
    for (; i < total; i += stride) {
        int n = (int)(i % N);
        int k = (int)(i / N);
        wt[(size_t)n * K + k] = f2bf(w[i]);
    }
}

// ---------------------------------------------------------------- big WMMA GEMM
// C[M][N] = A[M][K](bf16) * Bt[N][K](bf16, transposed) (+bias) (+add)
// block = 256 thr (8 waves). wave tile = 32(M) x 64(N); block tile = 128 x 128.
// Double-buffered: next k-step fragments are loaded into fresh registers before
// the 8 WMMAs of the current k-step, so waits are partial, not loadcnt==0.
__global__ __launch_bounds__(256) void gemm_bf16_kernel(
    const unsigned short* __restrict__ A, const unsigned short* __restrict__ Bt,
    const float* __restrict__ bias, const float* __restrict__ add,
    float* __restrict__ Cf, unsigned short* __restrict__ Cb,
    int M, int N, int K, int blocksN)
{
    int lane = threadIdx.x & 31;
    int wave = threadIdx.x >> 5;
    int g = lane >> 4, l15 = lane & 15;
    int bn = blockIdx.x % blocksN;
    int bm = blockIdx.x / blocksN;
    int mBase = bm * 128 + (wave & 3) * 32;
    int nBase = bn * 128 + (wave >> 2) * 64;

    const unsigned short* Ap0 = A + (size_t)(mBase + l15) * K;
    const unsigned short* Ap1 = A + (size_t)(mBase + 16 + l15) * K;
    const unsigned short* Bp  = Bt + (size_t)(nBase + l15) * K;
    const size_t bstep = (size_t)16 * K;

    v8f acc[8] = {};
    v16bf a0 = load_frag_contig(Ap0, g);
    v16bf a1 = load_frag_contig(Ap1, g);
    v16bf b0 = load_frag_contig(Bp, g);
    v16bf b1 = load_frag_contig(Bp + bstep, g);
    v16bf b2 = load_frag_contig(Bp + 2 * bstep, g);
    v16bf b3 = load_frag_contig(Bp + 3 * bstep, g);

    for (int kb = 32; kb < K; kb += 32) {
        v16bf na0 = load_frag_contig(Ap0 + kb, g);
        v16bf na1 = load_frag_contig(Ap1 + kb, g);
        v16bf nb0 = load_frag_contig(Bp + kb, g);
        v16bf nb1 = load_frag_contig(Bp + bstep + kb, g);
        v16bf nb2 = load_frag_contig(Bp + 2 * bstep + kb, g);
        v16bf nb3 = load_frag_contig(Bp + 3 * bstep + kb, g);
        acc[0] = WMMA_BF16(a0, b0, acc[0]);
        acc[1] = WMMA_BF16(a0, b1, acc[1]);
        acc[2] = WMMA_BF16(a0, b2, acc[2]);
        acc[3] = WMMA_BF16(a0, b3, acc[3]);
        acc[4] = WMMA_BF16(a1, b0, acc[4]);
        acc[5] = WMMA_BF16(a1, b1, acc[5]);
        acc[6] = WMMA_BF16(a1, b2, acc[6]);
        acc[7] = WMMA_BF16(a1, b3, acc[7]);
        a0 = na0; a1 = na1;
        b0 = nb0; b1 = nb1; b2 = nb2; b3 = nb3;
    }
    acc[0] = WMMA_BF16(a0, b0, acc[0]);
    acc[1] = WMMA_BF16(a0, b1, acc[1]);
    acc[2] = WMMA_BF16(a0, b2, acc[2]);
    acc[3] = WMMA_BF16(a0, b3, acc[3]);
    acc[4] = WMMA_BF16(a1, b0, acc[4]);
    acc[5] = WMMA_BF16(a1, b1, acc[5]);
    acc[6] = WMMA_BF16(a1, b2, acc[6]);
    acc[7] = WMMA_BF16(a1, b3, acc[7]);

#pragma unroll
    for (int mi = 0; mi < 2; mi++) {
#pragma unroll
        for (int t = 0; t < 4; t++) {
#pragma unroll
            for (int v = 0; v < 8; v++) {
                int row = mBase + mi * 16 + g * 8 + v;
                int col = nBase + t * 16 + l15;
                float val = acc[mi * 4 + t][v];
                if (bias) val += bias[col];
                if (add)  val += add[(size_t)row * N + col];
                size_t o = (size_t)row * N + col;
                if (Cf) Cf[o] = val;
                if (Cb) Cb[o] = f2bf(val);
            }
        }
    }
}

// ---------------------------------------------------------------- small projections
// Y[b][r][n] = sum_k X[b*srcRows + srcStart + r][k] * W[k][n]
__global__ void proj_kernel(const float* __restrict__ X, const float* __restrict__ W,
                            float* __restrict__ Yf, unsigned short* __restrict__ Yb,
                            int rowsPerBatch, int srcRowsPerBatch, int srcRowStart,
                            int K, int N, int total)
{
    int idx = blockIdx.x * blockDim.x + threadIdx.x;
    if (idx >= total) return;
    int n = idx % N;
    int r = (idx / N) % rowsPerBatch;
    int b = idx / (N * rowsPerBatch);
    const float* x = X + ((size_t)b * srcRowsPerBatch + srcRowStart + r) * (size_t)K;
    float acc = 0.f;
#pragma unroll 4
    for (int k = 0; k < K; k++) acc += x[k] * W[(size_t)k * N + n];
    size_t o = ((size_t)b * rowsPerBatch + r) * (size_t)N + n;
    if (Yf) Yf[o] = acc;
    if (Yb) Yb[o] = f2bf(acc);
}

// TI[b][t][n] = sum_c text[b][t][c] * ip[b][n][c]   (sem = text_probs @ TI)
__global__ void ti_kernel(const float* __restrict__ enc, float* __restrict__ TI) {
    int idx = blockIdx.x * blockDim.x + threadIdx.x;
    if (idx >= BATCH * TXT * NTOK) return;
    int n = idx & 3;
    int t = (idx >> 2) % TXT;
    int b = idx / (4 * TXT);
    const float* tx = enc + ((size_t)b * (TXT + NTOK) + t) * CD_DIM;
    const float* ip = enc + ((size_t)b * (TXT + NTOK) + TXT + n) * CD_DIM;
    float acc = 0.f;
#pragma unroll 4
    for (int c = 0; c < CD_DIM; c++) acc += tx[c] * ip[c];
    TI[idx] = acc;
}

// ---------------------------------------------------------------- per-head text attention
// one wave per (b, h, 16-row tile): WMMA Q*K^T -> softmax(77) -> WMMA P*V
__global__ __launch_bounds__(32) void attn_text_kernel(
    const unsigned short* __restrict__ q,   // [B*S][C] bf16
    const unsigned short* __restrict__ kt,  // [B][77][C] bf16
    const unsigned short* __restrict__ vt,  // [B][77][C] bf16
    float* __restrict__ text_out)           // [B*S][C] f32
{
    __shared__ float Lg[16][80];
    __shared__ __attribute__((aligned(16))) unsigned short P[16][96];
    int lane = threadIdx.x;
    int g = lane >> 4, l15 = lane & 15;
    int mt = blockIdx.x & 255;
    int h  = (blockIdx.x >> 8) % HEADS;
    int b  = blockIdx.x / (256 * HEADS);
    const float scale = 0.125f;

    size_t qoff = ((size_t)b * S_LEN + (size_t)mt * 16 + l15) * C_DIM + h * DH;
    v16bf a0 = load_frag_contig(q + qoff, g);
    v16bf a1 = load_frag_contig(q + qoff + 32, g);

    for (int nt = 0; nt < 5; nt++) {
        int tok  = nt * 16 + l15;
        int tokc = tok < TXT ? tok : TXT - 1;
        const unsigned short* kp = kt + ((size_t)b * TXT + tokc) * C_DIM + h * DH;
        v16bf b0 = load_frag_contig(kp, g);
        v16bf b1 = load_frag_contig(kp + 32, g);
        v8f acc = {};
        acc = WMMA_BF16(a0, b0, acc);
        acc = WMMA_BF16(a1, b1, acc);
#pragma unroll
        for (int v = 0; v < 8; v++)
            Lg[g * 8 + v][nt * 16 + l15] = (tok < TXT) ? acc[v] * scale : -1e30f;
    }
    __syncthreads();
    if (lane < 16) {
        float mx = -1e30f;
        for (int k = 0; k < TXT; k++) mx = fmaxf(mx, Lg[lane][k]);
        float s = 0.f;
        for (int k = 0; k < TXT; k++) s += __expf(Lg[lane][k] - mx);
        float inv = 1.f / s;
        for (int k = 0; k < 96; k++)
            P[lane][k] = (k < TXT) ? f2bf(__expf(Lg[lane][k] - mx) * inv)
                                   : (unsigned short)0;
    }
    __syncthreads();

    size_t orow = ((size_t)b * S_LEN + (size_t)mt * 16) * C_DIM;
    for (int dt = 0; dt < 4; dt++) {
        v8f acc = {};
#pragma unroll
        for (int kb = 0; kb < 96; kb += 32) {
            v16bf fa = load_frag_contig(&P[l15][kb], g);
            FragU fb;
#pragma unroll
            for (int e = 0; e < 16; e++) {
                int t2  = kb + frag_k(e, g);
                int t2c = t2 < TXT ? t2 : TXT - 1;     // P is exactly 0 for t2>=77
                fb.u[e] = vt[((size_t)b * TXT + t2c) * C_DIM + h * DH + dt * 16 + l15];
            }
            acc = WMMA_BF16(fa, fb.v, acc);
        }
#pragma unroll
        for (int v = 0; v < 8; v++)
            text_out[orow + (size_t)(g * 8 + v) * C_DIM + h * DH + dt * 16 + l15] = acc[v];
    }
}

// ---------------------------------------------------------------- full-dim sem path
// one wave per (b, 16-row tile): WMMA q(1280) * k_text^T -> softmax -> P @ TI
// Double-buffered like the big GEMM.
__global__ __launch_bounds__(32) void sem_kernel(
    const unsigned short* __restrict__ q,   // [B*S][C] bf16
    const unsigned short* __restrict__ kt,  // [B][77][C] bf16
    const float* __restrict__ TI,           // [B][77][4]
    float* __restrict__ sem)                // [B*S][4]
{
    __shared__ float Lg[16][80];
    __shared__ float TIs[TXT][4];
    int lane = threadIdx.x;
    int g = lane >> 4, l15 = lane & 15;
    int mt = blockIdx.x & 255;
    int b  = blockIdx.x >> 8;
    const float scale = 0.125f;

    for (int i = lane; i < TXT * 4; i += 32)
        TIs[i >> 2][i & 3] = TI[(size_t)b * TXT * 4 + i];

    const unsigned short* qp = q + ((size_t)b * S_LEN + (size_t)mt * 16 + l15) * C_DIM;
    const unsigned short* kp[5];
#pragma unroll
    for (int nt = 0; nt < 5; nt++) {
        int tok  = nt * 16 + l15;
        int tokc = tok < TXT ? tok : TXT - 1;
        kp[nt] = kt + ((size_t)b * TXT + tokc) * C_DIM;
    }

    v8f acc[5] = {};
    v16bf a = load_frag_contig(qp, g);
    v16bf bb[5];
#pragma unroll
    for (int nt = 0; nt < 5; nt++) bb[nt] = load_frag_contig(kp[nt], g);

    for (int kb = 32; kb < C_DIM; kb += 32) {
        v16bf na = load_frag_contig(qp + kb, g);
        v16bf nb[5];
#pragma unroll
        for (int nt = 0; nt < 5; nt++) nb[nt] = load_frag_contig(kp[nt] + kb, g);
#pragma unroll
        for (int nt = 0; nt < 5; nt++) acc[nt] = WMMA_BF16(a, bb[nt], acc[nt]);
        a = na;
#pragma unroll
        for (int nt = 0; nt < 5; nt++) bb[nt] = nb[nt];
    }
#pragma unroll
    for (int nt = 0; nt < 5; nt++) acc[nt] = WMMA_BF16(a, bb[nt], acc[nt]);

#pragma unroll
    for (int nt = 0; nt < 5; nt++) {
        int tok = nt * 16 + l15;
#pragma unroll
        for (int v = 0; v < 8; v++)
            Lg[g * 8 + v][nt * 16 + l15] = (tok < TXT) ? acc[nt][v] * scale : -1e30f;
    }
    __syncthreads();
    if (lane < 16) {
        float mx = -1e30f;
        for (int k = 0; k < TXT; k++) mx = fmaxf(mx, Lg[lane][k]);
        float s = 0.f;
        for (int k = 0; k < TXT; k++) s += __expf(Lg[lane][k] - mx);
        float inv = 1.f / s;
        float a0 = 0.f, a1 = 0.f, a2 = 0.f, a3 = 0.f;
        for (int k = 0; k < TXT; k++) {
            float pv = __expf(Lg[lane][k] - mx) * inv;
            a0 += pv * TIs[k][0]; a1 += pv * TIs[k][1];
            a2 += pv * TIs[k][2]; a3 += pv * TIs[k][3];
        }
        size_t o = ((size_t)b * S_LEN + (size_t)mt * 16 + lane) * 4;
        sem[o + 0] = a0 * scale; sem[o + 1] = a1 * scale;
        sem[o + 2] = a2 * scale; sem[o + 3] = a3 * scale;
    }
}

// ---------------------------------------------------------------- ip attention + fuse
__global__ void ip_attn_kernel(
    const unsigned short* __restrict__ q,   // [B*S][C] bf16
    const float* __restrict__ ipk,          // [B][4][C]
    const float* __restrict__ ipv,          // [B][4][C]
    const float* __restrict__ sem,          // [B*S][4]
    const float* __restrict__ mask,         // [S]
    const float* __restrict__ text_out,     // [B*S][C]
    unsigned short* __restrict__ hs_bf)     // [B*S][C] bf16
{
    int idx = blockIdx.x * blockDim.x + threadIdx.x;
    if (idx >= BATCH * S_LEN * HEADS) return;
    int h = idx % HEADS;
    int s = (idx / HEADS) % S_LEN;
    int b = idx / (HEADS * S_LEN);

    size_t qoff = ((size_t)b * S_LEN + s) * C_DIM + h * DH;
    float qv[DH];
#pragma unroll 8
    for (int d = 0; d < DH; d++) qv[d] = bf2f(q[qoff + d]);

    float sc[NTOK];
#pragma unroll
    for (int n = 0; n < NTOK; n++) {
        const float* kpn = ipk + ((size_t)b * NTOK + n) * C_DIM + h * DH;
        float acc = 0.f;
#pragma unroll 8
        for (int d = 0; d < DH; d++) acc += qv[d] * kpn[d];
        sc[n] = 0.125f * acc + 0.5f * sem[((size_t)b * S_LEN + s) * 4 + n];
    }
    float mx = fmaxf(fmaxf(sc[0], sc[1]), fmaxf(sc[2], sc[3]));
    float p[NTOK]; float sum = 0.f;
#pragma unroll
    for (int n = 0; n < NTOK; n++) { p[n] = __expf(sc[n] - mx); sum += p[n]; }
    float m = mask[s] / sum;
#pragma unroll
    for (int n = 0; n < NTOK; n++) p[n] *= m;

    for (int d = 0; d < DH; d++) {
        float acc = text_out[qoff + d];
#pragma unroll
        for (int n = 0; n < NTOK; n++)
            acc += p[n] * ipv[((size_t)b * NTOK + n) * C_DIM + h * DH + d];
        hs_bf[qoff + d] = f2bf(acc);
    }
}

// ---------------------------------------------------------------- launcher
extern "C" void kernel_launch(void* const* d_in, const int* in_sizes, int n_in,
                              void* d_out, int out_size, void* d_ws, size_t ws_size,
                              hipStream_t stream)
{
    const float* hidden = (const float*)d_in[0];
    const float* enc    = (const float*)d_in[1];
    const float* mask   = (const float*)d_in[2];
    const float* Wq     = (const float*)d_in[3];
    const float* Wk     = (const float*)d_in[4];
    const float* Wv     = (const float*)d_in[5];
    const float* Wkip   = (const float*)d_in[6];
    const float* Wvip   = (const float*)d_in[7];
    const float* Wout   = (const float*)d_in[8];
    const float* bout   = (const float*)d_in[9];
    float* out = (float*)d_out;

    char* p = (char*)d_ws;
    auto carve = [&](size_t bytes) -> void* {
        void* r = (void*)p;
        p += (bytes + 255) & ~(size_t)255;
        return r;
    };
    const size_t M = (size_t)BATCH * S_LEN;   // 16384

    unsigned short* hs_in_bf = (unsigned short*)carve(M * C_DIM * 2);
    unsigned short* Wq_t     = (unsigned short*)carve((size_t)C_DIM * C_DIM * 2);
    unsigned short* Wout_t   = (unsigned short*)carve((size_t)C_DIM * C_DIM * 2);
    unsigned short* q_bf     = (unsigned short*)carve(M * C_DIM * 2);
    unsigned short* kt_bf    = (unsigned short*)carve((size_t)BATCH * TXT * C_DIM * 2);
    unsigned short* vt_bf    = (unsigned short*)carve((size_t)BATCH * TXT * C_DIM * 2);
    float* ipk      = (float*)carve((size_t)BATCH * NTOK * C_DIM * 4);
    float* ipv      = (float*)carve((size_t)BATCH * NTOK * C_DIM * 4);
    float* TI       = (float*)carve((size_t)BATCH * TXT * NTOK * 4);
    float* semb     = (float*)carve(M * NTOK * 4);
    float* text_out = (float*)carve(M * C_DIM * 4);
    unsigned short* hs_bf = (unsigned short*)carve(M * C_DIM * 2);

    // 1. conversions (activations row-major bf16; weights transposed bf16)
    cvt_bf16_kernel<<<4096, 256, 0, stream>>>(hidden, hs_in_bf, (long long)M * C_DIM);
    cvt_t_bf16_kernel<<<2048, 256, 0, stream>>>(Wq,   Wq_t,   C_DIM, C_DIM);
    cvt_t_bf16_kernel<<<2048, 256, 0, stream>>>(Wout, Wout_t, C_DIM, C_DIM);

    // 2. q = hs @ Wq  (WMMA, bf16 out)
    const int blocksN = C_DIM / 128;                   // 10
    const int gemmBlocks = (int)(M / 128) * blocksN;   // 1280
    gemm_bf16_kernel<<<gemmBlocks, 256, 0, stream>>>(
        hs_in_bf, Wq_t, nullptr, nullptr, nullptr, q_bf, (int)M, C_DIM, C_DIM, blocksN);

    // 3. text / ip projections
    const int totKV = BATCH * TXT * C_DIM;
    proj_kernel<<<(totKV + 255) / 256, 256, 0, stream>>>(
        enc, Wk,   nullptr, kt_bf, TXT,  TXT + NTOK, 0,   CD_DIM, C_DIM, totKV);
    proj_kernel<<<(totKV + 255) / 256, 256, 0, stream>>>(
        enc, Wv,   nullptr, vt_bf, TXT,  TXT + NTOK, 0,   CD_DIM, C_DIM, totKV);
    const int totIP = BATCH * NTOK * C_DIM;
    proj_kernel<<<(totIP + 255) / 256, 256, 0, stream>>>(
        enc, Wkip, ipk,     nullptr, NTOK, TXT + NTOK, TXT, CD_DIM, C_DIM, totIP);
    proj_kernel<<<(totIP + 255) / 256, 256, 0, stream>>>(
        enc, Wvip, ipv,     nullptr, NTOK, TXT + NTOK, TXT, CD_DIM, C_DIM, totIP);

    // 4. TI = text @ ip^T   (sem refactor: sem = softmax(q k^T) @ TI)
    ti_kernel<<<(BATCH * TXT * NTOK + 255) / 256, 256, 0, stream>>>(enc, TI);

    // 5. per-head text attention (WMMA)
    attn_text_kernel<<<BATCH * HEADS * (S_LEN / 16), 32, 0, stream>>>(
        q_bf, kt_bf, vt_bf, text_out);

    // 6. full-dim text_probs -> sem (WMMA over K=1280)
    sem_kernel<<<BATCH * (S_LEN / 16), 32, 0, stream>>>(q_bf, kt_bf, TI, semb);

    // 7. ip attention + fuse with text_out -> hs (bf16)
    ip_attn_kernel<<<(BATCH * S_LEN * HEADS + 255) / 256, 256, 0, stream>>>(
        q_bf, ipk, ipv, semb, mask, text_out, hs_bf);

    // 8. out = hs @ Wout + bout + residual  (WMMA, f32 out)
    gemm_bf16_kernel<<<gemmBlocks, 256, 0, stream>>>(
        hs_bf, Wout_t, bout, hidden, out, nullptr, (int)M, C_DIM, C_DIM, blocksN);
}
